// Sparse2L_59176059404714
// MI455X (gfx1250) — matrix operand order repaired and checked by reading, not compiled, not run on hardware
//
#include <hip/hip_runtime.h>

// ---- problem constants (from reference) ----
constexpr int Bb = 512, Dd = 2048, Kk = 4096, Cc = 512;
constexpr int NSTEPS = 10;
#define LRc    0.001f
#define GAMMAc 0.1f

typedef __attribute__((ext_vector_type(16))) __bf16 v16bf;
typedef __attribute__((ext_vector_type(8)))  __bf16 bf16x8;
typedef __attribute__((ext_vector_type(8)))  float  v8f;
typedef unsigned int v4u __attribute__((ext_vector_type(4)));
typedef int v8i __attribute__((ext_vector_type(8)));
typedef int v4i __attribute__((ext_vector_type(4)));

// LDS tile row stride: 32 bf16 payload + 8 bf16 pad = 80 bytes (20 DWORDs).
// 20-DWORD stride => 16 fragment columns hit 16 disjoint bank quads: conflict-free.
constexpr int LDS_STRIDE = 40;

// fp32 -> bf16, round to nearest even
__device__ __forceinline__ __bf16 tobf(float f) {
  unsigned u = __builtin_bit_cast(unsigned, f);
  unsigned r = u + 0x7FFFu + ((u >> 16) & 1u);
  unsigned short h = (unsigned short)(r >> 16);
  return __builtin_bit_cast(__bf16, h);
}

__device__ __forceinline__ unsigned lds_addr32(const void* p) {
  return (unsigned)(unsigned long long)p;   // low 32 bits = LDS byte offset
}

// ---- TDM: DMA a (rows x 32) bf16 tile (row stride ld elements) into LDS with
// +16B padding per 64B row, per CDNA5 ISA D# layout.
__device__ __forceinline__ void tdm_load_tile(const __bf16* g, int ld,
                                              unsigned lds_byte_addr, int rows) {
  unsigned long long ga = (unsigned long long)g;
  v4u g0;
  g0[0] = 1u;                                          // count=1, user mode
  g0[1] = lds_byte_addr;                               // lds_addr
  g0[2] = (unsigned)(ga & 0xFFFFFFFFu);                // global_addr[31:0]
  g0[3] = (unsigned)((ga >> 32) & 0x1FFFFFFu)          // global_addr[56:32]
        | (2u << 30);                                  // type=2 ("image")
  v8i g1;
  g1[0] = (int)((1u << 16)      // data_size = 1 -> 2 bytes
              | (1u << 20)      // pad_enable
              | (3u << 22)      // pad_interval: 16 DWORDs (64B = one tile row)
              | (3u << 25));    // pad_amount: 4 DWORDs (16B)
  g1[1] = (int)(((unsigned)ld & 0xFFFFu) << 16);       // tensor_dim0[15:0]
  g1[2] = (int)((((unsigned)ld >> 16) & 0xFFFFu)       // tensor_dim0[31:16]
              | (((unsigned)rows & 0xFFFFu) << 16));   // tensor_dim1[15:0]
  g1[3] = (int)(32u << 16);                            // tensor_dim1 hi=0 | tile_dim0=32
  g1[4] = rows;                                        // tile_dim1=rows, tile_dim2=0
  g1[5] = ld;                                          // tensor_dim0_stride[31:0]
  g1[6] = 0;                                           // stride0 hi | stride1 lo
  g1[7] = 0;
  v4i z4 = {0, 0, 0, 0};
  v8i z8 = {0, 0, 0, 0, 0, 0, 0, 0};
  // clang-23 / therock-10.0 arity: (g0 v4u, g1 v8i, v4i, v4i, v8i, i32 cpol)
  __builtin_amdgcn_tensor_load_to_lds(g0, g1, z4, z4, z8, 0);
}

// ---- LDS fragment loads (padded tiles, local k range 0..31) ----
__device__ __forceinline__ v16bf lds_afrag(const __bf16* sA, int m0, int lane) {
  int row = m0 + (lane & 15);
  int kb  = (lane >> 4) << 3;                 // +8 for upper half-wave
  const __bf16* p = sA + row * LDS_STRIDE + kb;
  bf16x8 lo = *(const bf16x8*)(p);
  bf16x8 hi = *(const bf16x8*)(p + 16);
  v16bf r;
#pragma unroll
  for (int i = 0; i < 8; ++i) { r[i] = lo[i]; r[i + 8] = hi[i]; }
  return r;
}

__device__ __forceinline__ v16bf lds_bfrag(const __bf16* sB, int n0, int lane) {
  int col = n0 + (lane & 15);
  int kb  = (lane >> 4) << 3;                 // +8 for upper half-wave
  const __bf16* p = sB + col * LDS_STRIDE + kb;
  bf16x8 lo = *(const bf16x8*)(p);
  bf16x8 hi = *(const bf16x8*)(p + 16);
  v16bf r;
#pragma unroll
  for (int i = 0; i < 8; ++i) { r[i] = lo[i]; r[i + 8] = hi[i]; }
  return r;
}

__device__ __forceinline__ v8f wmma_bf16(v16bf a, v16bf b, v8f c) {
  return __builtin_amdgcn_wmma_f32_16x16x32_bf16(false, a, false, b, (short)0, c,
                                                 false, false);
}

// ---- TDM-staged, double-buffered block GEMM: 32x32 per wave, acc += A*B ----
// A block tile: 64 x 32 ; B block tile: BROWS x 32 (BT layout source).
template <int KD, int LDA, int LDB, int BROWS>
__device__ __forceinline__ void gemm_staged(const __bf16* __restrict__ Ag, int mblock,
                                            const __bf16* __restrict__ Bg, int nblock,
                                            __bf16* sA, __bf16* sB, v8f (&acc)[2][2],
                                            int w, int lane, int m0l, int n0l) {
  constexpr int NK  = KD / 32;
  constexpr int ASZ = 64 * LDS_STRIDE;       // elements per A buffer
  constexpr int BSZ = BROWS * LDS_STRIDE;    // elements per B buffer
  const __bf16* Abase = Ag + (size_t)mblock * LDA;
  const __bf16* Bbase = Bg + (size_t)nblock * LDB;
  __syncthreads();                           // LDS reuse across calls is safe
  if (w == 0) {
    tdm_load_tile(Abase, LDA, lds_addr32(sA), 64);
    tdm_load_tile(Bbase, LDB, lds_addr32(sB), BROWS);
  }
  for (int i = 0; i < NK; ++i) {
    int cur = i & 1;
    if (w == 0) __builtin_amdgcn_s_wait_tensorcnt(0);  // stage i landed
    __syncthreads();                                   // visible to all waves;
                                                       // all reads of buf cur^1 done
    if (w == 0 && (i + 1) < NK) {                      // prefetch next stage (DMA
      int nxt = cur ^ 1;                               // overlaps the WMMAs below)
      tdm_load_tile(Abase + (i + 1) * 32, LDA, lds_addr32(sA + nxt * ASZ), 64);
      tdm_load_tile(Bbase + (i + 1) * 32, LDB, lds_addr32(sB + nxt * BSZ), BROWS);
    }
    const __bf16* cA = sA + cur * ASZ;
    const __bf16* cB = sB + cur * BSZ;
    v16bf a0 = lds_afrag(cA, m0l, lane);
    v16bf a1 = lds_afrag(cA, m0l + 16, lane);
    v16bf b0 = lds_bfrag(cB, n0l, lane);
    v16bf b1 = lds_bfrag(cB, n0l + 16, lane);
    acc[0][0] = wmma_bf16(a0, b0, acc[0][0]);
    acc[0][1] = wmma_bf16(a0, b1, acc[0][1]);
    acc[1][0] = wmma_bf16(a1, b0, acc[1][0]);
    acc[1][1] = wmma_bf16(a1, b1, acc[1][1]);
  }
}

// ---------------- conversion / init kernels ----------------
__global__ void k_conv_W(const float* __restrict__ W, __bf16* __restrict__ Wbf,
                         __bf16* __restrict__ WT) {
  int i = blockIdx.x * blockDim.x + threadIdx.x;
  if (i >= Kk * Dd) return;
  int k = i / Dd, d = i % Dd;
  __bf16 v = tobf(W[i]);
  Wbf[i] = v;                       // (K,D): BT layout for grad_x = r @ W^T
  WT[(size_t)d * Kk + k] = v;       // (D,K): BT layout for out = x @ W
}

__global__ void k_conv_V(const float* __restrict__ V, __bf16* __restrict__ Vbf,
                         __bf16* __restrict__ VT) {
  int i = blockIdx.x * blockDim.x + threadIdx.x;
  if (i >= Cc * Kk) return;
  int c = i / Kk, k = i % Kk;
  __bf16 v = tobf(V[i]);
  Vbf[i] = v;                       // (C,K): BT layout for grad_u = g @ V^T
  VT[(size_t)k * Cc + c] = v;       // (K,C): BT layout for z = u @ V
}

__global__ void k_init_x(float* __restrict__ x, __bf16* __restrict__ xbf) {
  int i = blockIdx.x * blockDim.x + threadIdx.x;
  if (i >= Bb * Kk) return;
  x[i] = 0.0f;
  xbf[i] = tobf(0.0f);
}

__global__ void k_init_u(const float* __restrict__ u0, float* __restrict__ u,
                         __bf16* __restrict__ ubf) {
  int i = blockIdx.x * blockDim.x + threadIdx.x;
  if (i >= Bb * Cc) return;
  float v = u0[i];
  u[i] = v;
  ubf[i] = tobf(v);
}

// ---------------- step kernel 1: out = x @ W ; r = 2*(out - inputs) ----------------
// grid (Bb/64, Dd/128), block 256 (8 waves in 2x4), 32x32 per wave.
__global__ __launch_bounds__(256) void k_gemm_out(
    const __bf16* __restrict__ xbf, const __bf16* __restrict__ WT,
    const float* __restrict__ inputs, float* __restrict__ out,
    __bf16* __restrict__ rbf) {
  __shared__ __attribute__((aligned(64))) __bf16 sA[2 * 64 * LDS_STRIDE];
  __shared__ __attribute__((aligned(64))) __bf16 sB[2 * 128 * LDS_STRIDE];
  int lane = threadIdx.x & 31, w = threadIdx.x >> 5;
  int m0l = (w & 1) * 32, n0l = (w >> 1) * 32;
  int mblock = blockIdx.x * 64, nblock = blockIdx.y * 128;
  v8f acc[2][2] = {};
  gemm_staged<Kk, Kk, Kk, 128>(xbf, mblock, WT, nblock, sA, sB, acc, w, lane, m0l, n0l);
  int colo = lane & 15, rowo = (lane >> 4) * 8;
#pragma unroll
  for (int ti = 0; ti < 2; ++ti)
#pragma unroll
    for (int tj = 0; tj < 2; ++tj)
#pragma unroll
      for (int e = 0; e < 8; ++e) {
        int row = mblock + m0l + ti * 16 + rowo + e;
        int col = nblock + n0l + tj * 16 + colo;
        size_t idx = (size_t)row * Dd + col;
        float o = acc[ti][tj][e];
        out[idx] = o;
        rbf[idx] = tobf(2.0f * (o - inputs[idx]));
      }
}

// ---- step kernel 2: grad_x = r @ W^T ; z = u @ V ; fused x prox update ----
// grid (Bb/64, Kk/128), block 256, 32x32 per wave; two staged reductions.
__global__ __launch_bounds__(256) void k_gemm_xupd(
    const __bf16* __restrict__ rbf, const __bf16* __restrict__ Wbf,
    const __bf16* __restrict__ ubf, const __bf16* __restrict__ VT,
    float* __restrict__ x, __bf16* __restrict__ xbf, __bf16* __restrict__ gbf) {
  __shared__ __attribute__((aligned(64))) __bf16 sA[2 * 64 * LDS_STRIDE];
  __shared__ __attribute__((aligned(64))) __bf16 sB[2 * 128 * LDS_STRIDE];
  int lane = threadIdx.x & 31, w = threadIdx.x >> 5;
  int m0l = (w & 1) * 32, n0l = (w >> 1) * 32;
  int mblock = blockIdx.x * 64, nblock = blockIdx.y * 128;
  v8f ag[2][2] = {}, az[2][2] = {};
  gemm_staged<Dd, Dd, Dd, 128>(rbf, mblock, Wbf, nblock, sA, sB, ag, w, lane, m0l, n0l);
  gemm_staged<Cc, Cc, Cc, 128>(ubf, mblock, VT,  nblock, sA, sB, az, w, lane, m0l, n0l);
  int colo = lane & 15, rowo = (lane >> 4) * 8;
#pragma unroll
  for (int ti = 0; ti < 2; ++ti)
#pragma unroll
    for (int tj = 0; tj < 2; ++tj)
#pragma unroll
      for (int e = 0; e < 8; ++e) {
        int row = mblock + m0l + ti * 16 + rowo + e;
        int col = nblock + n0l + tj * 16 + colo;
        size_t idx = (size_t)row * Kk + col;
        float z  = az[ti][tj][e];
        float g  = ag[ti][tj][e];
        float ex = 0.5f * __expf(-z);                 // 0.5*exp(-z)
        float t  = LRc * GAMMAc * (0.5f + ex);        // lr*gamma*causes
        float a  = x[idx] - LRc * g;
        float xn = fmaxf(a - t, 0.0f) + fminf(a + t, 0.0f);  // prox
        x[idx]   = xn;
        xbf[idx] = tobf(xn);
        gbf[idx] = tobf(-ex * fabsf(xn) * GAMMAc);    // dcost_dz
      }
}

// ---- step kernel 3: grad_u = g @ V^T + 0.02*gamma*u ; fused u prox update ----
// grid (Bb/64, Cc/64), block 128 (4 waves in 2x2), 32x32 per wave.
__global__ __launch_bounds__(128) void k_gemm_uupd(
    const __bf16* __restrict__ gbf, const __bf16* __restrict__ Vbf,
    float* __restrict__ u, __bf16* __restrict__ ubf) {
  __shared__ __attribute__((aligned(64))) __bf16 sA[2 * 64 * LDS_STRIDE];
  __shared__ __attribute__((aligned(64))) __bf16 sB[2 * 64 * LDS_STRIDE];
  int lane = threadIdx.x & 31, w = threadIdx.x >> 5;
  int m0l = (w & 1) * 32, n0l = (w >> 1) * 32;
  int mblock = blockIdx.x * 64, nblock = blockIdx.y * 64;
  v8f acc[2][2] = {};
  gemm_staged<Kk, Kk, Kk, 64>(gbf, mblock, Vbf, nblock, sA, sB, acc, w, lane, m0l, n0l);
  int colo = lane & 15, rowo = (lane >> 4) * 8;
#pragma unroll
  for (int ti = 0; ti < 2; ++ti)
#pragma unroll
    for (int tj = 0; tj < 2; ++tj)
#pragma unroll
      for (int e = 0; e < 8; ++e) {
        int row = mblock + m0l + ti * 16 + rowo + e;
        int col = nblock + n0l + tj * 16 + colo;
        size_t idx = (size_t)row * Cc + col;
        float uv = u[idx];
        float gu = acc[ti][tj][e] + 0.02f * GAMMAc * uv;
        float a  = uv - LRc * gu;
        float un = fmaxf(a - LRc * GAMMAc, 0.0f) + fminf(a + LRc * GAMMAc, 0.0f);
        u[idx]   = un;
        ubf[idx] = tobf(un);
      }
}

// ---------------- host launcher ----------------
extern "C" void kernel_launch(void* const* d_in, const int* in_sizes, int n_in,
                              void* d_out, int out_size, void* d_ws, size_t ws_size,
                              hipStream_t stream) {
  (void)in_sizes; (void)n_in; (void)out_size; (void)ws_size;
  const float* inputs = (const float*)d_in[0];   // (B,D)
  const float* W      = (const float*)d_in[1];   // (K,D)
  const float* V      = (const float*)d_in[2];   // (C,K)
  const float* u0     = (const float*)d_in[3];   // (B,C)
  float* out = (float*)d_out;                    // (B,D)

  char* ws = (char*)d_ws;
  size_t off = 0;
  auto alloc = [&](size_t bytes) -> char* {
    char* p = ws + off;
    off += (bytes + 255) & ~(size_t)255;
    return p;
  };
  __bf16* Wbf = (__bf16*)alloc((size_t)Kk * Dd * 2);  // (K,D)
  __bf16* WT  = (__bf16*)alloc((size_t)Dd * Kk * 2);  // (D,K)
  __bf16* Vbf = (__bf16*)alloc((size_t)Cc * Kk * 2);  // (C,K)
  __bf16* VT  = (__bf16*)alloc((size_t)Kk * Cc * 2);  // (K,C)
  float*  x   = (float*) alloc((size_t)Bb * Kk * 4);
  __bf16* xbf = (__bf16*)alloc((size_t)Bb * Kk * 2);
  float*  u   = (float*) alloc((size_t)Bb * Cc * 4);
  __bf16* ubf = (__bf16*)alloc((size_t)Bb * Cc * 2);
  __bf16* rbf = (__bf16*)alloc((size_t)Bb * Dd * 2);
  __bf16* gbf = (__bf16*)alloc((size_t)Bb * Kk * 2);

  // One-time operand conversion + state init (every call; deterministic).
  k_conv_W<<<(Kk * Dd + 255) / 256, 256, 0, stream>>>(W, Wbf, WT);
  k_conv_V<<<(Cc * Kk + 255) / 256, 256, 0, stream>>>(V, Vbf, VT);
  k_init_x<<<(Bb * Kk + 255) / 256, 256, 0, stream>>>(x, xbf);
  k_init_u<<<(Bb * Cc + 255) / 256, 256, 0, stream>>>(u0, u, ubf);

  for (int s = 0; s < NSTEPS; ++s) {
    k_gemm_out <<<dim3(Bb / 64, Dd / 128), 256, 0, stream>>>(xbf, WT, inputs, out, rbf);
    k_gemm_xupd<<<dim3(Bb / 64, Kk / 128), 256, 0, stream>>>(rbf, Wbf, ubf, VT, x, xbf, gbf);
    k_gemm_uupd<<<dim3(Bb / 64, Cc / 64), 128, 0, stream>>>(gbf, Vbf, u, ubf);
  }
}
